// LRLSAttention_87703232184479
// MI455X (gfx1250) — compile-verified, matrix operand order checked by reading, not run on hardware
//
#include <hip/hip_runtime.h>
#include <hip/hip_bf16.h>

typedef _Float16 half8 __attribute__((ext_vector_type(8)));
typedef _Float16 v16h  __attribute__((ext_vector_type(16)));
typedef float    v8f   __attribute__((ext_vector_type(8)));
typedef int      i4v   __attribute__((ext_vector_type(4)));

#define WMMA_F32_F16(a, b, c) \
  __builtin_amdgcn_wmma_f32_16x16x32_f16(false, (a), false, (b), (short)0, (c), false, false)

#define AS1 __attribute__((address_space(1)))
#define AS3 __attribute__((address_space(3)))

// Async global->LDS copy (CDNA5 GLOBAL_LOAD_ASYNC_TO_LDS_B128, ASYNCcnt),
// with a synchronous fallback if the builtin is unavailable.
// Probe-verified: the builtin's params are (global int4*, local int4*, imm, imm).
__device__ __forceinline__ void async_copy_b128(const _Float16* g, _Float16* l) {
#if __has_builtin(__builtin_amdgcn_global_load_async_to_lds_b128)
  __builtin_amdgcn_global_load_async_to_lds_b128(
      (AS1 i4v*)g, (AS3 i4v*)l, 0, 0);
#else
  *(half8*)l = *(const half8*)g;
#endif
}
__device__ __forceinline__ void wait_async() {
#if __has_builtin(__builtin_amdgcn_global_load_async_to_lds_b128)
#if __has_builtin(__builtin_amdgcn_s_wait_asynccnt)
  __builtin_amdgcn_s_wait_asynccnt(0);
#else
  asm volatile("s_wait_asynccnt 0x0" ::: "memory");
#endif
#endif
}

// Problem dims (match reference)
constexpr int BB  = 4;
constexpr int TT  = 2048;
constexpr int DD  = 512;
constexpr int HH  = 8;
constexpr int KDM = 64;
constexpr int DFF = 2048;
constexpr int BT  = BB * TT;          // 8192 rows
constexpr int NQT = TT / 16;          // 128 query tiles per (b,h)

// ---------------------------------------------------------------------------
// fp32 -> fp16 cast (weights)
// ---------------------------------------------------------------------------
__global__ void cvt_f32_f16(const float* __restrict__ src, _Float16* __restrict__ dst, int n) {
  int i = blockIdx.x * blockDim.x + threadIdx.x;
  if (i < n) dst[i] = (_Float16)src[i];
}

// ---------------------------------------------------------------------------
// LayerNorm: one wave (32 lanes) per row of D=512, writes fp16
// ---------------------------------------------------------------------------
__global__ __launch_bounds__(256)
void ln_rows(const float* __restrict__ x, const float* __restrict__ g,
             const float* __restrict__ b, _Float16* __restrict__ out, int nrows) {
  int wave = (int)((blockIdx.x * blockDim.x + threadIdx.x) >> 5);
  int lane = threadIdx.x & 31;
  if (wave >= nrows) return;
  const float* row = x + (size_t)wave * DD;
  float vals[16];
  float s = 0.f;
#pragma unroll
  for (int i = 0; i < 16; ++i) { vals[i] = row[lane + i * 32]; s += vals[i]; }
#pragma unroll
  for (int off = 1; off < 32; off <<= 1) s += __shfl_xor(s, off, 32);
  float mean = s * (1.0f / DD);
  float vs = 0.f;
#pragma unroll
  for (int i = 0; i < 16; ++i) { float d = vals[i] - mean; vs += d * d; }
#pragma unroll
  for (int off = 1; off < 32; off <<= 1) vs += __shfl_xor(vs, off, 32);
  float rstd = rsqrtf(vs * (1.0f / DD) + 1e-6f);
  _Float16* orow = out + (size_t)wave * DD;
#pragma unroll
  for (int i = 0; i < 16; ++i) {
    int c = lane + i * 32;
    orow[c] = (_Float16)((vals[i] - mean) * rstd * g[c] + b[c]);
  }
}

// ---------------------------------------------------------------------------
// Generic fp16 WMMA GEMM: C[M,N] = A[M,K] @ B[K,N] (+bias)(+relu)(+resid)
// Block tile 256(M) x 64(N), 8 waves; each wave owns a 32x64 strip:
// 2 A-fragments x 4 B-fragments = 8 WMMAs per K-step.
// A staged via async global->LDS B128; B staged transposed for contiguous
// 16B B-fragment ds loads.
// ---------------------------------------------------------------------------
__global__ __launch_bounds__(256)
void gemm_wmma(const _Float16* __restrict__ A, const _Float16* __restrict__ Bw,
               const float* __restrict__ bias, const float* __restrict__ resid,
               _Float16* __restrict__ out16, float* __restrict__ out32,
               int M, int N, int K, int doRelu) {
  constexpr int LDA = 40;   // halves; 80B row stride (16B aligned)
  constexpr int LDB = 40;
  __shared__ _Float16 As[256 * LDA];  // 20.0 KB
  __shared__ _Float16 Bs[64 * LDB];   // 5.0 KB, transposed: [n][k]

  int tid  = threadIdx.x;
  int w    = tid >> 5;
  int lane = tid & 31;
  int mBase = blockIdx.y * 256;
  int nBase = blockIdx.x * 64;
  int mloc  = lane & 15;
  int lh    = lane >> 4;

  v8f acc[2][4];
#pragma unroll
  for (int mi = 0; mi < 2; ++mi)
#pragma unroll
    for (int nt = 0; nt < 4; ++nt)
#pragma unroll
      for (int i = 0; i < 8; ++i) acc[mi][nt][i] = 0.f;

  for (int k0 = 0; k0 < K; k0 += 32) {
    // --- stage A tile: 256 rows x 32 halves; thread t copies row t (async) ---
    {
      const _Float16* src = A + (size_t)(mBase + tid) * K + k0;
      if (k0 + 32 < K) __builtin_prefetch(src + 32, 0, 3);
#pragma unroll
      for (int c = 0; c < 4; ++c)
        async_copy_b128(src + c * 8, &As[tid * LDA + c * 8]);
    }
    // --- stage B tile transposed: 32(k) x 64(n) -> Bs[n][k] ---
    {
      int kr = tid >> 3, n0 = (tid & 7) * 8;
      const _Float16* src = Bw + (size_t)(k0 + kr) * N + nBase + n0;
      half8 p = *(const half8*)(src);
#pragma unroll
      for (int j = 0; j < 8; ++j) Bs[(n0 + j) * LDB + kr] = p[j];
    }
    wait_async();
    __syncthreads();

    // --- A fragments (16x32, ISA 16-bit layout), rows w*32 + mi*16 ---
    v16h af[2];
#pragma unroll
    for (int mi = 0; mi < 2; ++mi) {
      int rloc = w * 32 + mi * 16 + mloc;
      half8 lo = *(const half8*)(&As[rloc * LDA + lh * 8]);
      half8 hi = *(const half8*)(&As[rloc * LDA + 16 + lh * 8]);
#pragma unroll
      for (int i = 0; i < 8; ++i) { af[mi][i] = lo[i]; af[mi][8 + i] = hi[i]; }
    }
    // --- 4 B fragments, 8 WMMAs ---
#pragma unroll
    for (int nt = 0; nt < 4; ++nt) {
      int n = nt * 16 + mloc;
      v16h bf;
      half8 lo = *(const half8*)(&Bs[n * LDB + lh * 8]);
      half8 hi = *(const half8*)(&Bs[n * LDB + 16 + lh * 8]);
#pragma unroll
      for (int i = 0; i < 8; ++i) { bf[i] = lo[i]; bf[8 + i] = hi[i]; }
      acc[0][nt] = WMMA_F32_F16(af[0], bf, acc[0][nt]);
      acc[1][nt] = WMMA_F32_F16(af[1], bf, acc[1][nt]);
    }
    __syncthreads();
  }

  // --- epilogue: C layout lane(col)=L%16, row = r + 8*(L/16) ---
  int cn = lane & 15, rh = lane >> 4;
#pragma unroll
  for (int mi = 0; mi < 2; ++mi)
#pragma unroll
    for (int nt = 0; nt < 4; ++nt) {
      int gn = nBase + nt * 16 + cn;
      float bv = bias ? bias[gn] : 0.f;
#pragma unroll
      for (int r = 0; r < 8; ++r) {
        int gm = mBase + w * 32 + mi * 16 + 8 * rh + r;
        float vv = acc[mi][nt][r] + bv;
        if (doRelu) vv = fmaxf(vv, 0.f);
        if (resid) vv += resid[(size_t)gm * N + gn];
        if (out32) out32[(size_t)gm * N + gn] = vv;
        else       out16[(size_t)gm * N + gn] = (_Float16)vv;
      }
    }
}

// ---------------------------------------------------------------------------
// Block-sparse flash attention: one wave per 16-row query tile of one (b,h).
// Visits key tile 0 (globals) + local tiles qt-2..qt+2; tile 0 query rows
// (global rows, i<16) scan all key tiles. Online softmax, KD=64.
// ---------------------------------------------------------------------------
__global__ __launch_bounds__(128)
void attn_wmma(const _Float16* __restrict__ Q, const _Float16* __restrict__ Kt,
               const _Float16* __restrict__ V, _Float16* __restrict__ ctx) {
  __shared__ _Float16 Pl[4][16 * 16];   // per-wave P staging (C->A layout)

  int w    = threadIdx.x >> 5;
  int lane = threadIdx.x & 31;
  int qidx = blockIdx.x * 4 + w;                 // global query-tile id
  int b    = qidx / (HH * NQT);
  int rem  = qidx % (HH * NQT);
  int h    = rem / NQT;
  int qt   = rem % NQT;
  int mloc = lane & 15;
  int lh   = lane >> 4;

  const size_t rs = (size_t)HH * KDM;            // 512 halves per token row
  const size_t hoff = (size_t)h * KDM;
  const _Float16* qb = Q  + (size_t)(b * TT) * rs + hoff;
  const _Float16* kb = Kt + (size_t)(b * TT) * rs + hoff;
  const _Float16* vb = V  + (size_t)(b * TT) * rs + hoff;

  // Q fragments: 16x64 -> two 16x32 A-fragments
  v16h qf[2];
#pragma unroll
  for (int ks = 0; ks < 2; ++ks) {
    const _Float16* base = qb + (size_t)(qt * 16 + mloc) * rs + ks * 32;
    half8 lo = *(const half8*)(base + lh * 8);
    half8 hi = *(const half8*)(base + 16 + lh * 8);
#pragma unroll
    for (int i = 0; i < 8; ++i) { qf[ks][i] = lo[i]; qf[ks][8 + i] = hi[i]; }
  }

  v8f O[4];
  float mrow[8], lrow[8];
#pragma unroll
  for (int nt = 0; nt < 4; ++nt)
#pragma unroll
    for (int i = 0; i < 8; ++i) O[nt][i] = 0.f;
#pragma unroll
  for (int r = 0; r < 8; ++r) { mrow[r] = -3.0e38f; lrow[r] = 0.f; }

  auto process = [&](int st, bool needMask) {
    int sbase = st * 16;
    // scores S = Q K^T / sqrt(KD)
    v8f S;
#pragma unroll
    for (int i = 0; i < 8; ++i) S[i] = 0.f;
    {
      const _Float16* krow = kb + (size_t)(sbase + mloc) * rs;
#pragma unroll
      for (int ks = 0; ks < 2; ++ks) {
        v16h bf;
        half8 lo = *(const half8*)(krow + ks * 32 + lh * 8);
        half8 hi = *(const half8*)(krow + ks * 32 + 16 + lh * 8);
#pragma unroll
        for (int i = 0; i < 8; ++i) { bf[i] = lo[i]; bf[8 + i] = hi[i]; }
        S = WMMA_F32_F16(qf[ks], bf, S);
      }
    }
#pragma unroll
    for (int r = 0; r < 8; ++r) S[r] *= 0.125f;
    if (needMask) {
      int j = sbase + mloc;
#pragma unroll
      for (int r = 0; r < 8; ++r) {
        int i  = qt * 16 + 8 * lh + r;
        int di = i - j; if (di < 0) di = -di;
        bool ok = (di <= 32) || (j < 16) || (i < 16);
        if (!ok) S[r] = -1.0e9f;
      }
    }
    // online softmax (rows live across 16-lane halves)
    float pexp[8];
#pragma unroll
    for (int r = 0; r < 8; ++r) {
      float rm = S[r];
#pragma unroll
      for (int off = 1; off < 16; off <<= 1) rm = fmaxf(rm, __shfl_xor(rm, off, 32));
      float mn    = fmaxf(mrow[r], rm);
      float alpha = __expf(mrow[r] - mn);
      float p     = __expf(S[r] - mn);
      float sum   = p;
#pragma unroll
      for (int off = 1; off < 16; off <<= 1) sum += __shfl_xor(sum, off, 32);
      lrow[r] = lrow[r] * alpha + sum;
      mrow[r] = mn;
#pragma unroll
      for (int nt = 0; nt < 4; ++nt) O[nt][r] *= alpha;
      pexp[r] = p;
    }
    // stage P through LDS to convert C-layout -> A-layout (upper K half = 0)
#pragma unroll
    for (int r = 0; r < 8; ++r)
      Pl[w][(8 * lh + r) * 16 + mloc] = (_Float16)pexp[r];
    v16h ap;
    {
      half8 lo = *(const half8*)(&Pl[w][mloc * 16 + lh * 8]);
#pragma unroll
      for (int i = 0; i < 8; ++i) { ap[i] = lo[i]; ap[8 + i] = (_Float16)0.f; }
    }
    // O += P x V  (V tile is 16(s) x 64(kd); upper K half zero)
#pragma unroll
    for (int nt = 0; nt < 4; ++nt) {
      v16h bvf;
#pragma unroll
      for (int i = 0; i < 8; ++i) {
        bvf[i]     = vb[(size_t)(sbase + lh * 8 + i) * rs + nt * 16 + mloc];
        bvf[8 + i] = (_Float16)0.f;
      }
      O[nt] = WMMA_F32_F16(ap, bvf, O[nt]);
    }
  };

  if (qt == 0) {
    for (int st = 0; st < NQT; ++st) process(st, false);
  } else {
    process(0, false);                       // global columns (j < 16)
    int lo = qt - 2 < 1 ? 1 : qt - 2;
    int hi = qt + 2 > NQT - 1 ? NQT - 1 : qt + 2;
    for (int st = lo; st <= hi; ++st) process(st, true);
  }

  // normalize + write ctx (fp16, [B,T,H,KD])
  _Float16* cb = ctx + (size_t)(b * TT + qt * 16) * rs + hoff;
#pragma unroll
  for (int nt = 0; nt < 4; ++nt)
#pragma unroll
    for (int r = 0; r < 8; ++r) {
      float vv = O[nt][r] / lrow[r];
      cb[(size_t)(8 * lh + r) * rs + nt * 16 + mloc] = (_Float16)vv;
    }
}

// ---------------------------------------------------------------------------
// Host-side orchestration
// ---------------------------------------------------------------------------
extern "C" void kernel_launch(void* const* d_in, const int* in_sizes, int n_in,
                              void* d_out, int out_size, void* d_ws, size_t ws_size,
                              hipStream_t stream) {
  const float* x    = (const float*)d_in[0];
  const float* ln1g = (const float*)d_in[1];
  const float* ln1b = (const float*)d_in[2];
  const float* Wq   = (const float*)d_in[3];
  const float* bq   = (const float*)d_in[4];
  const float* Wk   = (const float*)d_in[5];
  const float* bk   = (const float*)d_in[6];
  const float* Wv   = (const float*)d_in[7];
  const float* bv   = (const float*)d_in[8];
  const float* Wo   = (const float*)d_in[9];
  const float* bo   = (const float*)d_in[10];
  const float* ln2g = (const float*)d_in[11];
  const float* ln2b = (const float*)d_in[12];
  const float* W1   = (const float*)d_in[13];
  const float* b1   = (const float*)d_in[14];
  const float* W2   = (const float*)d_in[15];
  const float* b2   = (const float*)d_in[16];
  float* out = (float*)d_out;

  char* ws = (char*)d_ws;
  const size_t MB = 1024ull * 1024ull;
  // Region 0 (32MB): xn16|q16|k16|v16 ; later aliased by FFN hidden h16
  _Float16* xn16 = (_Float16*)(ws + 0 * MB);
  _Float16* q16  = (_Float16*)(ws + 8 * MB);
  _Float16* k16  = (_Float16*)(ws + 16 * MB);
  _Float16* v16  = (_Float16*)(ws + 24 * MB);
  _Float16* h16  = (_Float16*)(ws + 0 * MB);    // alias (32MB), after attn
  _Float16* ctx16 = (_Float16*)(ws + 32 * MB);  // 8MB
  _Float16* xn2   = (_Float16*)(ws + 32 * MB);  // alias ctx16 after out-proj
  float*    x1    = (float*)   (ws + 40 * MB);  // 16MB
  _Float16* wq16 = (_Float16*)(ws + 56 * MB);
  _Float16* wk16 = (_Float16*)(ws + 57 * MB);
  _Float16* wv16 = (_Float16*)(ws + 58 * MB);
  _Float16* wo16 = (_Float16*)(ws + 59 * MB);
  _Float16* w1_16 = (_Float16*)(ws + 60 * MB);  // 2MB
  _Float16* w2_16 = (_Float16*)(ws + 62 * MB);  // 2MB

  const int WSQ = DD * HH * KDM;   // 512*512
  const int WS1 = DD * DFF;        // 512*2048
  auto cvt = [&](const float* s, _Float16* d, int n) {
    cvt_f32_f16<<<(n + 255) / 256, 256, 0, stream>>>(s, d, n);
  };
  cvt(Wq, wq16, WSQ); cvt(Wk, wk16, WSQ); cvt(Wv, wv16, WSQ);
  cvt(Wo, wo16, WSQ); cvt(W1, w1_16, WS1); cvt(W2, w2_16, WS1);

  // LN1 -> xn16
  ln_rows<<<BT / 8, 256, 0, stream>>>(x, ln1g, ln1b, xn16, BT);

  // QKV projections (fp16 out)
  dim3 gq(DD / 64, BT / 256);
  gemm_wmma<<<gq, 256, 0, stream>>>(xn16, wq16, bq, nullptr, q16, nullptr, BT, DD, DD, 0);
  gemm_wmma<<<gq, 256, 0, stream>>>(xn16, wk16, bk, nullptr, k16, nullptr, BT, DD, DD, 0);
  gemm_wmma<<<gq, 256, 0, stream>>>(xn16, wv16, bv, nullptr, v16, nullptr, BT, DD, DD, 0);

  // Block-sparse flash attention -> ctx16
  attn_wmma<<<(BB * HH * NQT) / 4, 128, 0, stream>>>(q16, k16, v16, ctx16);

  // Out projection + residual (fp32 x1)
  gemm_wmma<<<gq, 256, 0, stream>>>(ctx16, wo16, bo, x, nullptr, x1, BT, DD, DD, 0);

  // LN2 -> xn2
  ln_rows<<<BT / 8, 256, 0, stream>>>(x1, ln2g, ln2b, xn2, BT);

  // FFN1: relu(xn2 @ W1 + b1) -> h16
  dim3 g1(DFF / 64, BT / 256);
  gemm_wmma<<<g1, 256, 0, stream>>>(xn2, w1_16, b1, nullptr, h16, nullptr, BT, DFF, DD, 1);

  // FFN2: x1 + (h16 @ W2 + b2) -> out (fp32)
  dim3 g2(DD / 64, BT / 256);
  gemm_wmma<<<g2, 256, 0, stream>>>(h16, w2_16, b2, x1, nullptr, out, BT, DD, DFF, 0);
}
